// GraphAttentionLayer_19335942766875
// MI455X (gfx1250) — matrix operand order repaired and checked by reading, not compiled, not run on hardware
//
#include <hip/hip_runtime.h>
#include <hip/hip_bf16.h>

typedef __attribute__((ext_vector_type(16))) __bf16 v16bf;
typedef __attribute__((ext_vector_type(8)))  float  v8f;

#define BB 8
#define NN 1024
#define INF_ 128
#define OUTF 64
#define HH 4
#define BH (BB*HH)          // 32

// ---------------------------------------------------------------------------
// Prep kernels: x -> bf16 (row-major), W -> bf16 transposed Wt[n][k]
// so GEMM fragments become contiguous 16B loads.
// ---------------------------------------------------------------------------
__global__ __launch_bounds__(256) void k_prep_x(const float* __restrict__ x,
                                                __bf16* __restrict__ xbf) {
    int idx = blockIdx.x * 256 + threadIdx.x;       // 1048576
    xbf[idx] = (__bf16)x[idx];
}

__global__ __launch_bounds__(256) void k_prep_w(const float* __restrict__ W,
                                                __bf16* __restrict__ wbf) {
    int idx = blockIdx.x * 256 + threadIdx.x;       // 32768
    int n = idx >> 7;                                // 0..255 (col of W)
    int k = idx & 127;                               // 0..127 (row of W)
    wbf[n * INF_ + k] = (__bf16)W[(size_t)k * (OUTF * HH) + n];
}

// ---------------------------------------------------------------------------
// Kernel 1: h = x @ W   (8192x128) * (128x256), bf16 WMMA, f32 accum.
// Output stored transposed & bf16: hbf[(bh*64+f)*1024 + n]  (f-major) so the
// aggregation GEMM's B-fragments are contiguous along K=j.
// ---------------------------------------------------------------------------
__global__ __launch_bounds__(256) void k_gemm_h(const __bf16* __restrict__ xbf,
                                                const __bf16* __restrict__ wbf,
                                                __bf16* __restrict__ hbf) {
    const int lane = threadIdx.x & 31;
    const int wv   = threadIdx.x >> 5;
    const int task = blockIdx.x * 8 + wv;      // 2048 tasks
    const int rowTile = task >> 2;             // 0..511  (16 rows each)
    const int head    = task & 3;              // 64-col group == head
    const int m16 = lane & 15;
    const int grp = lane >> 4;
    const int row = rowTile * 16 + m16;        // global row for A-frag

    v8f acc[4] = {};

    for (int kt = 0; kt < INF_; kt += 32) {
        union { float4 q[2]; v16bf v; } ua;
        const __bf16* xr = xbf + (size_t)row * INF_ + kt + grp * 8;
        ua.q[0] = *(const float4*)xr;          // K = base+0..7
        ua.q[1] = *(const float4*)(xr + 16);   // K = base+16..23
        const int kb = kt + grp * 8;
#pragma unroll
        for (int nt = 0; nt < 4; ++nt) {
            const int ncol = head * 64 + nt * 16 + m16;
            const __bf16* wp = wbf + (size_t)ncol * INF_ + kb;
            union { float4 q[2]; v16bf v; } ub;
            ub.q[0] = *(const float4*)wp;
            ub.q[1] = *(const float4*)(wp + 16);
            acc[nt] = __builtin_amdgcn_wmma_f32_16x16x32_bf16(
                false, ua.v, false, ub.v, (short)0, acc[nt], false, false);
        }
    }

#pragma unroll
    for (int nt = 0; nt < 4; ++nt) {
        const int f = nt * 16 + m16;                    // D col = lane%16
#pragma unroll
        for (int r = 0; r < 8; ++r) {
            int grow = rowTile * 16 + r + grp * 8;      // D row
            int b  = grow >> 10;
            int n  = grow & 1023;
            int bh = b * HH + head;
            hbf[((size_t)bh * OUTF + f) * NN + n] = (__bf16)acc[nt][r];
        }
    }
}

// ---------------------------------------------------------------------------
// Kernel 2a: e_src/e_dst per (bh, n):  dot(h[bh,n,:], a_src/a_dst[head])
// ---------------------------------------------------------------------------
__global__ __launch_bounds__(256) void k_edots(const __bf16* __restrict__ hbf,
                                               const float* __restrict__ a,
                                               float* __restrict__ esrc,
                                               float* __restrict__ edst) {
    const int idx  = blockIdx.x * 256 + threadIdx.x;    // 32768
    const int bh   = idx >> 10;
    const int n    = idx & 1023;
    const int head = bh & 3;
    float es = 0.f, ed = 0.f;
#pragma unroll 4
    for (int f = 0; f < OUTF; ++f) {
        float hv = (float)hbf[((size_t)bh * OUTF + f) * NN + n];
        es += hv * a[head * 2 * OUTF + f];
        ed += hv * a[head * 2 * OUTF + OUTF + f];
    }
    esrc[idx] = es;
    edst[idx] = ed;
}

// ---------------------------------------------------------------------------
// Kernel 2b: row softmax stats.  One wave per (b,i), all 4 heads processed
// from a single coalesced adj-row read.  Online max/sum + shuffle reduce.
// ---------------------------------------------------------------------------
__global__ __launch_bounds__(256) void k_rowstats(const int* __restrict__ adj,
                                                  const float* __restrict__ esrc,
                                                  const float* __restrict__ edst,
                                                  float* __restrict__ rowm,
                                                  float* __restrict__ rinv) {
    const int lane = threadIdx.x & 31;
    const int wv   = threadIdx.x >> 5;
    const int row  = blockIdx.x * 8 + wv;               // 0..8191 == (b,i)
    const int b = row >> 10;
    const int i = row & 1023;
    const int* ar = adj + (size_t)row * NN;

    float es[HH], mx[HH], sm[HH];
#pragma unroll
    for (int h = 0; h < HH; ++h) {
        es[h] = esrc[(size_t)(b * HH + h) * NN + i];
        mx[h] = -3.0e38f;
        sm[h] = 0.f;
    }
    for (int j = lane; j < NN; j += 32) {
        if (ar[j] != 0) {
#pragma unroll
            for (int h = 0; h < HH; ++h) {
                float v = es[h] + edst[(size_t)(b * HH + h) * NN + j];
                v = v > 0.f ? v : 0.2f * v;             // LeakyReLU
                float nm = fmaxf(mx[h], v);
                sm[h] = sm[h] * __expf(mx[h] - nm) + __expf(v - nm);
                mx[h] = nm;
            }
        }
    }
#pragma unroll
    for (int off = 16; off > 0; off >>= 1) {
#pragma unroll
        for (int h = 0; h < HH; ++h) {
            float om = __shfl_xor(mx[h], off, 32);
            float os = __shfl_xor(sm[h], off, 32);
            float nm = fmaxf(mx[h], om);
            sm[h] = sm[h] * __expf(mx[h] - nm) + os * __expf(om - nm);
            mx[h] = nm;
        }
    }
    if (lane == 0) {
#pragma unroll
        for (int h = 0; h < HH; ++h) {
            size_t o = (size_t)(b * HH + h) * NN + i;
            rowm[o] = mx[h];
            rinv[o] = sm[h] > 0.f ? 1.f / sm[h] : 0.f;  // fully-masked row -> 0
        }
    }
}

// ---------------------------------------------------------------------------
// Kernel 3: out = softmax(P) @ h, P regenerated on the fly.
// Block = (b, 16-row i-tile); 4 waves = 4 heads share one LDS adj tile.
// adj tiles (16x128) are DOUBLE-BUFFERED in LDS and filled with
// GLOBAL_LOAD_ASYNC_TO_LDS_B128 (ASYNCcnt) so HBM latency overlaps WMMA.
// One barrier per 128-column stage.
// ---------------------------------------------------------------------------
// LDS layout (single flat object => offset 0 in LDS):
//   [0      , 16384)  : e_dst for 4 heads (float[4096])
//   [16384  , 24576)  : adj tile buffer 0 (int[2048])
//   [24576  , 32768)  : adj tile buffer 1 (int[2048])
#define LDSE_BYTES  (HH * NN * 4)           // 16384
#define ADJ_OFF     ((unsigned)LDSE_BYTES)  // 16384
#define ADJ_BYTES   (16 * 128 * 4)          // 8192

__global__ __launch_bounds__(128) void k_attn(const int* __restrict__ adj,
                                              const __bf16* __restrict__ hbf,
                                              const float* __restrict__ esrc,
                                              const float* __restrict__ edst,
                                              const float* __restrict__ rowm,
                                              const float* __restrict__ rinv,
                                              float* __restrict__ out) {
    __shared__ __align__(16) char smem[LDSE_BYTES + 2 * ADJ_BYTES];
    float* ldsE = (float*)smem;

    const int t    = threadIdx.x;
    const int lane = t & 31;
    const int head = t >> 5;            // wave id == head
    const int b    = blockIdx.x >> 6;   // 64 i-tiles per batch
    const int it   = blockIdx.x & 63;
    const int i0   = it * 16;

    // stage e_dst for all 4 heads of this batch (coalesced)
    for (int k = t; k < HH * NN; k += 128)
        ldsE[k] = edst[(size_t)b * HH * NN + k];

    const int m16 = lane & 15;
    const int grp = lane >> 4;
    const int bh  = b * HH + head;
    const int irow = i0 + m16;
    const float es_i  = esrc[(size_t)bh * NN + irow];
    const float m_i   = rowm[(size_t)bh * NN + irow];
    const float inv_i = rinv[(size_t)bh * NN + irow];

    v8f acc[4] = {};
    const int* adjBase = adj + ((size_t)b * NN + i0) * NN;

    // each thread stages 4 x 16B of the 16x128 adj tile, contiguous per row
    const int sr  = t >> 3;             // 0..15  (tile row)
    const int sc0 = (t & 7) * 4;        // 0..28  (int index within 32-col grp)

    // async-issue one 16x128 adj tile into LDS buffer at byte dstOff
    auto stage_async = [&](int jc, unsigned dstOff) {
#pragma unroll
        for (int q = 0; q < 4; ++q) {
            int c = sc0 + q * 32;
            unsigned loff = dstOff + (unsigned)(sr * 128 + c) * 4u;
            const int* gp = adjBase + (size_t)sr * NN + jc + c;
            asm volatile("global_load_async_to_lds_b128 %0, %1, off"
                         :: "v"(loff), "v"(gp) : "memory");
        }
    };

    stage_async(0, ADJ_OFF);            // prefetch tile 0 into buffer 0
    unsigned bufOff = 0;

    for (int j0 = 0; j0 < NN; j0 += 128) {
        asm volatile("s_wait_asynccnt 0x0" ::: "memory");
        __syncthreads();                // tile j0 resident; prev reads done
        if (j0 + 128 < NN)
            stage_async(j0 + 128, ADJ_OFF + (bufOff ^ ADJ_BYTES));

        const int* ldsA = (const int*)(smem + ADJ_OFF + bufOff);

        for (int jj = 0; jj < 4; ++jj) {
            const int jbase = jj * 32 + grp * 8;    // this lane's kl base
            int av_[16];
            float ev_[16];
            {
                const int* ap = &ldsA[m16 * 128 + jbase];
                *(int4*)&av_[0]  = *(const int4*)ap;
                *(int4*)&av_[4]  = *(const int4*)(ap + 4);
                *(int4*)&av_[8]  = *(const int4*)(ap + 16);
                *(int4*)&av_[12] = *(const int4*)(ap + 20);
                const float* ep = &ldsE[head * NN + j0 + jbase];
                *(float4*)&ev_[0]  = *(const float4*)ep;
                *(float4*)&ev_[4]  = *(const float4*)(ep + 4);
                *(float4*)&ev_[8]  = *(const float4*)(ep + 16);
                *(float4*)&ev_[12] = *(const float4*)(ep + 20);
            }
            v16bf af;
#pragma unroll
            for (int e = 0; e < 16; ++e) {
                float p = 0.f;
                if (av_[e] != 0) {
                    float v = es_i + ev_[e];
                    v = v > 0.f ? v : 0.2f * v;         // LeakyReLU
                    p = __expf(v - m_i) * inv_i;        // softmax prob
                }
                af[e] = (__bf16)p;
            }
            const int jcol = j0 + jj * 32;
#pragma unroll
            for (int nt = 0; nt < 4; ++nt) {
                const __bf16* hp = hbf +
                    ((size_t)bh * OUTF + nt * 16 + m16) * NN + jcol + grp * 8;
                union { float4 q[2]; v16bf v; } u;
                u.q[0] = *(const float4*)hp;
                u.q[1] = *(const float4*)(hp + 16);
                acc[nt] = __builtin_amdgcn_wmma_f32_16x16x32_bf16(
                    false, af, false, u.v, (short)0, acc[nt], false, false);
            }
        }
        bufOff ^= ADJ_BYTES;
    }

    // ---- write out[b][i][head*64+f], coalesced across lanes ----
#pragma unroll
    for (int nt = 0; nt < 4; ++nt) {
        const int f = head * OUTF + nt * 16 + m16;
#pragma unroll
        for (int r = 0; r < 8; ++r) {
            int i = i0 + r + grp * 8;
            out[((size_t)b * NN + i) * (HH * OUTF) + f] = acc[nt][r];
        }
    }
}

// ---------------------------------------------------------------------------
extern "C" void kernel_launch(void* const* d_in, const int* in_sizes, int n_in,
                              void* d_out, int out_size, void* d_ws, size_t ws_size,
                              hipStream_t stream) {
    const float* x   = (const float*)d_in[0];   // (8,1024,128)
    const int*   adj = (const int*)d_in[1];     // (8,1024,1024)
    const float* W   = (const float*)d_in[2];   // (128,256)
    const float* a   = (const float*)d_in[3];   // (4,128)
    float* out = (float*)d_out;                 // (8,1024,256)

    // workspace layout
    __bf16* hbf = (__bf16*)d_ws;                                     // 4 MB
    float*  esrc = (float*)((char*)d_ws + (size_t)4 * 1024 * 1024);  // 128 KB
    float*  edst = esrc + BH * NN;
    float*  rowm = edst + BH * NN;
    float*  rinv = rowm + BH * NN;
    __bf16* xbf  = (__bf16*)((char*)d_ws + (size_t)4 * 1024 * 1024 + 512 * 1024); // 2 MB
    __bf16* wbf  = xbf + (size_t)BB * NN * INF_;                     // 64 KB

    k_prep_x  <<<4096, 256, 0, stream>>>(x, xbf);
    k_prep_w  <<<128,  256, 0, stream>>>(W, wbf);
    k_gemm_h  <<<256,  256, 0, stream>>>(xbf, wbf, hbf);
    k_edots   <<<128,  256, 0, stream>>>(hbf, a, esrc, edst);
    k_rowstats<<<1024, 256, 0, stream>>>(adj, esrc, edst, rowm, rinv);
    k_attn    <<<512,  128, 0, stream>>>(adj, hbf, esrc, edst, rowm, rinv, out);
}
